// GraphEncoderNetwork_120259085246
// MI455X (gfx1250) — matrix operand order repaired and checked by reading, not compiled, not run on hardware
//
#include <hip/hip_runtime.h>
#include <hip/hip_bf16.h>

typedef __attribute__((ext_vector_type(16))) _Float16 v16h;
typedef __attribute__((ext_vector_type(8)))  float    v8f;

#define NW 4  // waves (tiles) per block for WMMA tile kernels

__device__ __forceinline__ int lane_id() { return threadIdx.x & 31; }

__device__ __forceinline__ v8f wmma_f16(v16h a, v16h b, v8f c) {
  // (neg_a, A, neg_b, B, c_mod, C, reuse_a, reuse_b)
  return __builtin_amdgcn_wmma_f32_16x16x32_f16(false, a, false, b, (short)0, c,
                                                false, false);
}

// ---- WMMA fragment builders (wave32 layouts per CDNA5 ISA 7.12.2) -------

// A fragment: 16x32 (MxK) f16, source = row-major f32 LDS tile [16][LD],
// columns [kbase, kbase+32). VGPR v: k = (v&3)*2 + half*8 + (v>>2)*16.
template <int LD>
__device__ __forceinline__ v16h frag_A(const float* s, int kbase, int kvalid) {
  const int lane = lane_id();
  const int m = lane & 15, half = lane >> 4;
  v16h a;
#pragma unroll
  for (int v = 0; v < 8; ++v) {
    int k = kbase + (v & 3) * 2 + half * 8 + (v >> 2) * 16;
    float f0 = (k < kvalid) ? s[m * LD + k] : 0.0f;
    float f1 = (k + 1 < kvalid) ? s[m * LD + k + 1] : 0.0f;
    a[2 * v] = (_Float16)f0;
    a[2 * v + 1] = (_Float16)f1;
  }
  return a;
}

// B fragment: 32x16 (KxN) f16 from row-major f32 weight W[K][LDW], col block nb.
// VGPR v: k = (v&3)*2 + half*8 + (v>>2)*16 (same K order as frag_A).
template <int LDW>
__device__ __forceinline__ v16h frag_B(const float* W, int nb, int nvalid,
                                       int kbase, int kvalid) {
  const int lane = lane_id();
  const int n = lane & 15, half = lane >> 4;
  const bool nv = (n < nvalid);
  v16h b;
#pragma unroll
  for (int v = 0; v < 8; ++v) {
    int k = kbase + (v & 3) * 2 + half * 8 + (v >> 2) * 16;
    float f0 = (nv && k < kvalid) ? W[k * LDW + nb + n] : 0.0f;
    float f1 = (nv && k + 1 < kvalid) ? W[(k + 1) * LDW + nb + n] : 0.0f;
    b[2 * v] = (_Float16)f0;
    b[2 * v + 1] = (_Float16)f1;
  }
  return b;
}

// Store 16x16 f32 C/D tile to row-major LDS [16][LDO] at col nb (+bias, relu).
// D layout: VGPR v -> row m = v + half*8, col n = lane&15.
template <int LDO>
__device__ __forceinline__ void store_D(float* o, int nb, v8f d,
                                        const float* bias, int nvalid,
                                        bool relu) {
  const int lane = lane_id();
  const int n = lane & 15, half = lane >> 4;
#pragma unroll
  for (int v = 0; v < 8; ++v) {
    int m = v + half * 8;
    float val = 0.0f;
    if (n < nvalid) {
      val = d[v] + bias[nb + n];
      if (relu) val = fmaxf(val, 0.0f);
    }
    o[m * LDO + nb + n] = val;
  }
}

// ---- one 16-row tile through a 3-layer MLP: [KIN]->16(relu)->8(relu)->32 ----
template <int KIN, int LDIN>
__device__ __forceinline__ void mlp_tile(const float* in, const float* W1,
                                         const float* b1, const float* W2,
                                         const float* b2, const float* W3,
                                         const float* b3, float* h,
                                         float* out) {
  // layer 1: [16xKIN] @ [KINx16]
  v8f c1 = {};
#pragma unroll
  for (int k0 = 0; k0 < KIN; k0 += 32)
    c1 = wmma_f16(frag_A<LDIN>(in, k0, KIN), frag_B<16>(W1, 0, 16, k0, KIN), c1);
  store_D<16>(h, 0, c1, b1, 16, true);
  __builtin_amdgcn_wave_barrier();
  // layer 2: [16x16] @ [16x8] (K padded to 32, N padded to 16)
  v8f c2 = wmma_f16(frag_A<16>(h, 0, 16), frag_B<8>(W2, 0, 8, 0, 16), (v8f){});
  __builtin_amdgcn_wave_barrier();
  store_D<16>(h, 0, c2, b2, 8, true);  // cols 8..15 zeroed
  __builtin_amdgcn_wave_barrier();
  // layer 3: [16x8] @ [8x32] -> two N-tiles (no relu)
  v16h a3 = frag_A<16>(h, 0, 8);
  v8f d0 = wmma_f16(a3, frag_B<32>(W3, 0, 16, 0, 8), (v8f){});
  v8f d1 = wmma_f16(a3, frag_B<32>(W3, 16, 16, 0, 8), (v8f){});
  store_D<32>(out, 0, d0, b3, 16, false);
  store_D<32>(out, 16, d1, b3 + 16, 16, false);
  __builtin_amdgcn_wave_barrier();
}

// -------------------- kernels --------------------

__global__ void zero_kernel(float* a, long n) {
  long i = (long)blockIdx.x * blockDim.x + threadIdx.x;
  long stride = (long)gridDim.x * blockDim.x;
  for (; i < n; i += stride) a[i] = 0.0f;
}

// fused prep + proc MLPs, one wave per 16 nodes
__global__ void __launch_bounds__(NW * 32) prep_proc_kernel(
    const float* __restrict__ x, const float* pW1, const float* pb1,
    const float* pW2, const float* pb2, const float* pW3, const float* pb3,
    const float* cW1, const float* cb1, const float* cW2, const float* cb2,
    const float* cW3, const float* cb3, float* __restrict__ x_prep,
    float* __restrict__ proc_out, int N) {
  __shared__ float s_xin[NW][16 * 64];
  __shared__ float s_h[NW][16 * 16];
  __shared__ float s_t1[NW][16 * 32];
  __shared__ float s_t2[NW][16 * 32];
  const int w = threadIdx.x >> 5;
  const int lane = threadIdx.x & 31;
  const int n0 = (blockIdx.x * NW + w) * 16;
  if (n0 >= N) return;
  for (int i = lane; i < 16 * 64; i += 32) {
    int r = i >> 6, c = i & 63;
    int node = n0 + r;
    s_xin[w][i] = (node < N) ? x[(long)node * 64 + c] : 0.0f;
  }
  __builtin_amdgcn_wave_barrier();
  mlp_tile<64, 64>(s_xin[w], pW1, pb1, pW2, pb2, pW3, pb3, s_h[w], s_t1[w]);
  for (int i = lane; i < 16 * 32; i += 32) {
    int r = i >> 5, c = i & 31;
    int node = n0 + r;
    if (node < N) x_prep[(long)node * 32 + c] = s_t1[w][i];
  }
  mlp_tile<32, 32>(s_t1[w], cW1, cb1, cW2, cb2, cW3, cb3, s_h[w], s_t2[w]);
  for (int i = lane; i < 16 * 32; i += 32) {
    int r = i >> 5, c = i & 31;
    int node = n0 + r;
    if (node < N) proc_out[(long)node * 32 + c] = s_t2[w][i];
  }
}

// x_agg[dst] += proc_out[src]; 8 threads/edge, float4 gather + f32 atomics
__global__ void edge_scatter_kernel(const int* __restrict__ ei,
                                    const float* __restrict__ proc_out,
                                    float* __restrict__ x_agg, int E) {
  long tid = (long)blockIdx.x * blockDim.x + threadIdx.x;
  if (tid >= (long)E * 8) return;
  int e = (int)(tid >> 3);
  int q = (int)(tid & 7);
  int src = ei[e];
  int dst = ei[E + e];
  const float4 v = reinterpret_cast<const float4*>(proc_out)[(long)src * 8 + q];
  float* base = x_agg + (long)dst * 32 + q * 4;
  atomicAdd(base + 0, v.x);
  atomicAdd(base + 1, v.y);
  atomicAdd(base + 2, v.z);
  atomicAdd(base + 3, v.w);
}

// node_emb = x_prep + agg(x_agg); merged = node([x, node_emb]); dag += merged
__global__ void __launch_bounds__(NW * 32) node_update_kernel(
    const float* __restrict__ x, const float* __restrict__ x_prep,
    const float* __restrict__ x_agg, const int* __restrict__ batch,
    const float* aW1, const float* ab1, const float* aW2, const float* ab2,
    const float* aW3, const float* ab3, const float* nW1, const float* nb1,
    const float* nW2, const float* nb2, const float* nW3, const float* nb3,
    float* __restrict__ node_emb_out, float* __restrict__ dag_emb_out, int N) {
  __shared__ float s_xin[NW][16 * 96];
  __shared__ float s_ain[NW][16 * 32];
  __shared__ float s_h[NW][16 * 16];
  __shared__ float s_t1[NW][16 * 32];
  const int w = threadIdx.x >> 5;
  const int lane = threadIdx.x & 31;
  const int n0 = (blockIdx.x * NW + w) * 16;
  if (n0 >= N) return;
  for (int i = lane; i < 16 * 32; i += 32) {
    int r = i >> 5, c = i & 31;
    int node = n0 + r;
    s_ain[w][i] = (node < N) ? x_agg[(long)node * 32 + c] : 0.0f;
  }
  __builtin_amdgcn_wave_barrier();
  mlp_tile<32, 32>(s_ain[w], aW1, ab1, aW2, ab2, aW3, ab3, s_h[w], s_t1[w]);
  for (int i = lane; i < 16 * 32; i += 32) {
    int r = i >> 5, c = i & 31;
    int node = n0 + r;
    float ne = 0.0f;
    if (node < N) {
      ne = x_prep[(long)node * 32 + c] + s_t1[w][i];
      node_emb_out[(long)node * 32 + c] = ne;
    }
    s_xin[w][r * 96 + 64 + c] = ne;
  }
  for (int i = lane; i < 16 * 64; i += 32) {
    int r = i >> 6, c = i & 63;
    int node = n0 + r;
    s_xin[w][r * 96 + c] = (node < N) ? x[(long)node * 64 + c] : 0.0f;
  }
  __builtin_amdgcn_wave_barrier();
  mlp_tile<96, 96>(s_xin[w], nW1, nb1, nW2, nb2, nW3, nb3, s_h[w], s_t1[w]);
  for (int i = lane; i < 16 * 32; i += 32) {
    int r = i >> 5, c = i & 31;
    int node = n0 + r;
    if (node < N) {
      int g = batch[node];
      atomicAdd(&dag_emb_out[(long)g * 32 + c], s_t1[w][i]);
    }
  }
}

// dag_p = dag(dag_emb); z[searchsorted(obs_indptr, g, right)-1] += dag_p
__global__ void __launch_bounds__(NW * 32) dag_kernel(
    const float* __restrict__ dag_emb, const int* __restrict__ obs_indptr,
    int n_ptr, const float* dW1, const float* db1, const float* dW2,
    const float* db2, const float* dW3, const float* db3,
    float* __restrict__ z_out, int G) {
  __shared__ float s_in[NW][16 * 32];
  __shared__ float s_h[NW][16 * 16];
  __shared__ float s_t1[NW][16 * 32];
  const int w = threadIdx.x >> 5;
  const int lane = threadIdx.x & 31;
  const int g0 = (blockIdx.x * NW + w) * 16;
  if (g0 >= G) return;
  for (int i = lane; i < 16 * 32; i += 32) {
    int r = i >> 5, c = i & 31;
    int g = g0 + r;
    s_in[w][i] = (g < G) ? dag_emb[(long)g * 32 + c] : 0.0f;
  }
  __builtin_amdgcn_wave_barrier();
  mlp_tile<32, 32>(s_in[w], dW1, db1, dW2, db2, dW3, db3, s_h[w], s_t1[w]);
  for (int i = lane; i < 16 * 32; i += 32) {
    int r = i >> 5, c = i & 31;
    int g = g0 + r;
    if (g < G) {
      int lo = 0, hi = n_ptr;  // searchsorted(obs_indptr, g, 'right') - 1
      while (lo < hi) {
        int mid = (lo + hi) >> 1;
        if (obs_indptr[mid] <= g) lo = mid + 1; else hi = mid;
      }
      int oid = lo - 1;
      atomicAdd(&z_out[(long)oid * 32 + c], s_t1[w][i]);
    }
  }
}

extern "C" void kernel_launch(void* const* d_in, const int* in_sizes, int n_in,
                              void* d_out, int out_size, void* d_ws,
                              size_t ws_size, hipStream_t stream) {
  const float* x = (const float*)d_in[0];
  const int* edge_index = (const int*)d_in[1];
  const int* batch = (const int*)d_in[2];
  const int* obs_indptr = (const int*)d_in[3];
  const float* P[30];
  for (int i = 0; i < 30; ++i) P[i] = (const float*)d_in[4 + i];
  // P[0..5]=prep, [6..11]=proc, [12..17]=agg, [18..23]=node, [24..29]=dag

  const int N = in_sizes[0] / 64;       // F = 64
  const int E = in_sizes[1] / 2;
  const int OBS = in_sizes[3] - 1;
  const int G = out_size / 32 - N - OBS;  // out = [N*32 | G*32 | OBS*32]

  float* x_prep = (float*)d_ws;
  float* proc_o = x_prep + (size_t)N * 32;
  float* x_agg = proc_o + (size_t)N * 32;
  float* node_o = (float*)d_out;
  float* dag_o = node_o + (size_t)N * 32;
  float* z_o = dag_o + (size_t)G * 32;

  zero_kernel<<<1024, 256, 0, stream>>>(x_agg, (long)N * 32);
  zero_kernel<<<64, 256, 0, stream>>>(dag_o, (long)(G + OBS) * 32);

  const int tiles = (N + 15) / 16;
  const int blocks = (tiles + NW - 1) / NW;
  prep_proc_kernel<<<blocks, NW * 32, 0, stream>>>(
      x, P[0], P[1], P[2], P[3], P[4], P[5], P[6], P[7], P[8], P[9], P[10],
      P[11], x_prep, proc_o, N);

  long ethreads = (long)E * 8;
  int eblocks = (int)((ethreads + 255) / 256);
  edge_scatter_kernel<<<eblocks, 256, 0, stream>>>(edge_index, proc_o, x_agg, E);

  node_update_kernel<<<blocks, NW * 32, 0, stream>>>(
      x, x_prep, x_agg, batch, P[12], P[13], P[14], P[15], P[16], P[17], P[18],
      P[19], P[20], P[21], P[22], P[23], node_o, dag_o, N);

  const int gtiles = (G + 15) / 16;
  const int gblocks = (gtiles + NW - 1) / NW;
  dag_kernel<<<gblocks, NW * 32, 0, stream>>>(dag_o, obs_indptr, OBS + 1, P[24],
                                              P[25], P[26], P[27], P[28], P[29],
                                              z_o, G);
}
